// VoxelLevel_77764677862183
// MI455X (gfx1250) — compile-verified, matrix operand order checked by reading, not run on hardware
//
#include <hip/hip_runtime.h>
#include <hip/hip_bf16.h>
#include <math.h>

// ---------------- CDNA5 WMMA / TDM types ----------------
typedef __attribute__((ext_vector_type(16))) __bf16        v16bf;
typedef __attribute__((ext_vector_type(8)))  float         v8f;
typedef __attribute__((ext_vector_type(4)))  unsigned int  u32x4v;
typedef __attribute__((ext_vector_type(8)))  int           i32x8v;
typedef __attribute__((ext_vector_type(4)))  int           i32x4v;

#if defined(__has_builtin)
#  if __has_builtin(__builtin_amdgcn_tensor_load_to_lds)
#    define USE_TDM 1
#  endif
#endif
#ifndef USE_TDM
#  define USE_TDM 0
#endif

#define RR 32
#define V32 32768
#define V16 4096
#define V8  512
#define NB  2
#define NPTS 65536

typedef unsigned short u16;

__device__ __forceinline__ u16 f2bf(float f) {
    unsigned int u = __float_as_uint(f);
    unsigned int r = u + 0x7FFFu + ((u >> 16) & 1u);   // RNE
    return (u16)(r >> 16);
}
__device__ __forceinline__ float bf2f(u16 h) {
    return __uint_as_float(((unsigned int)h) << 16);
}

// ---------------- zero ----------------
__global__ void k_zero(float* p, long n) {
    long i = (long)blockIdx.x * blockDim.x + threadIdx.x;
    if (i < n) p[i] = 0.f;
}

// ---------------- weight f32 -> bf16 [Cout][Ktot] ----------------
__global__ void k_cvt(const float* __restrict__ src, u16* __restrict__ dst, long n) {
    long i = (long)blockIdx.x * blockDim.x + threadIdx.x;
    if (i < n) dst[i] = f2bf(src[i]);
}

// ---------------- scatter-mean accumulate ----------------
__global__ void k_scatter(const float* __restrict__ feats, const float* __restrict__ pos,
                          float* __restrict__ voxsum, float* __restrict__ cnt) {
    long i = (long)blockIdx.x * blockDim.x + threadIdx.x;
    if (i >= (long)NB * NPTS) return;
    int b = (int)(i / NPTS), n = (int)(i % NPTS);
    const float* pp = pos + ((long)b * NPTS + n) * 3;
    int ix = min(max((int)(pp[0] * RR), 0), RR - 1);
    int iy = min(max((int)(pp[1] * RR), 0), RR - 1);
    int iz = min(max((int)(pp[2] * RR), 0), RR - 1);
    int flat = ix * RR * RR + iy * RR + iz;
    atomicAdd(cnt + (long)b * V32 + flat, 1.f);
    const float* fp = feats + ((long)b * NPTS + n) * 64;
    float* vs = voxsum + (long)b * 64 * V32 + flat;
    for (int c = 0; c < 64; ++c) atomicAdd(vs + (long)c * V32, fp[c]);
}

// ---------------- context channels ----------------
__global__ void k_ctx(const float* __restrict__ widths, float* __restrict__ ctx) {
    int b = blockIdx.x;
    __shared__ float wd[3][RR], lw[3][RR], cen[3][RR];
    if (threadIdx.x < 3) {
        int a = threadIdx.x;
        float run = 0.f;
        for (int i = 0; i < RR; ++i) {
            float w = widths[(b * 3 + a) * RR + i];
            wd[a][i] = w;
            lw[a][i] = logf(fmaxf(w, 1e-5f));
            cen[a][i] = run + 0.5f * w;
            run += w;
        }
    }
    __syncthreads();
    for (int v = threadIdx.x; v < V32; v += blockDim.x) {
        int x = v % RR, y = (v / RR) % RR, z = v / (RR * RR);
        float vol = wd[0][x] * wd[1][y] * wd[2][z];
        float* c = ctx + (long)b * 7 * V32 + v;
        c[0 * V32] = lw[0][x];
        c[1 * V32] = lw[1][y];
        c[2 * V32] = lw[2][z];
        c[3 * V32] = logf(fmaxf(vol, 1e-5f));
        c[4 * V32] = cen[0][x];
        c[5 * V32] = cen[1][y];
        c[6 * V32] = cen[2][z];
    }
}

// ---------------- fuse -> bf16 x0 ----------------
__global__ void k_fuse(const float* __restrict__ voxsum, const float* __restrict__ cnt,
                       const float* __restrict__ ctx, const float* __restrict__ fw,
                       const float* __restrict__ fb, u16* __restrict__ x0) {
    long i = (long)blockIdx.x * blockDim.x + threadIdx.x;
    if (i >= (long)NB * 64 * V32) return;
    int v = (int)(i % V32), o = (int)((i / V32) % 64), b = (int)(i / (64L * V32));
    float inv = 1.f / fmaxf(cnt[(long)b * V32 + v], 1.f);
    float s = fb[o];
    const float* vs = voxsum + (long)b * 64 * V32 + v;
    const float* w = fw + o * 71;
    for (int c = 0; c < 64; ++c) s += w[c] * vs[(long)c * V32] * inv;
    const float* cx = ctx + (long)b * 7 * V32 + v;
    for (int j = 0; j < 7; ++j) s += w[64 + j] * cx[(long)j * V32];
    x0[i] = f2bf(s);
}

// ---------------- 3x3x3 conv: implicit GEMM, bf16 WMMA, TDM weight stage ----------------
// Tile: 16 (Cout) x 128 (voxels). 128 threads = 4 waves, 2 WMMA accumulators/wave.
__global__ void k_conv3(const u16* __restrict__ in1, const u16* __restrict__ in2,
                        int cin1, int cin2, const u16* __restrict__ wb,
                        const float* __restrict__ bias, float* __restrict__ out,
                        int Din, int Cout) {
    const int numVox = Din * Din * Din;
    const int Cin = cin1 + cin2;
    const int Ktot = Cin * 27;
    const int b = blockIdx.z;
    const int coutBase = blockIdx.y * 16;
    const int voxBase = blockIdx.x * 128;
    const int tid = threadIdx.x;
    const int wv = tid >> 5, lane = tid & 31;
    const int half = (lane >= 16) ? 8 : 0;
    const int l15 = lane & 15;

    __shared__ __align__(16) u16 sA[16 * 32];    // [m][k]
    __shared__ __align__(16) u16 sBt[128 * 32];  // [n][k] (transposed for b128 frag loads)

    v8f acc0 = {0.f, 0.f, 0.f, 0.f, 0.f, 0.f, 0.f, 0.f};
    v8f acc1 = {0.f, 0.f, 0.f, 0.f, 0.f, 0.f, 0.f, 0.f};

#if USE_TDM
    unsigned ldsA = (unsigned)(unsigned long long)(void*)sA;
#endif

    for (int kk = 0; kk < Ktot; kk += 32) {
#if USE_TDM
        // --- Tensor Data Mover: async copy 16x32 bf16 weight tile into LDS ---
        if (tid < 32) {
            unsigned long long ga =
                (unsigned long long)(const void*)(wb + ((long)coutBase * Ktot + kk));
            u32x4v g0;
            g0[0] = 1u;                                     // count=1, user descriptor
            g0[1] = ldsA;                                   // lds_addr
            g0[2] = (unsigned)ga;                           // global_addr[31:0]
            g0[3] = (unsigned)(ga >> 32) | (2u << 30);      // global_addr[56:32] | type=2
            i32x8v g1;
            g1[0] = 0x00010000;                             // data_size=1 (2 bytes)
            g1[1] = (int)(((unsigned)Ktot & 0xffffu) << 16);            // dim0[15:0]
            g1[2] = (int)((((unsigned)Ktot >> 16) & 0xffffu) |
                          (((unsigned)Cout & 0xffffu) << 16));          // dim0 hi | dim1 lo
            g1[3] = (int)(32u << 16);                       // dim1 hi=0 | tile_dim0=32
            g1[4] = 16;                                     // tile_dim1=16, tile_dim2=0
            g1[5] = (int)Ktot;                              // tensor_dim0_stride lo32
            g1[6] = 0;
            g1[7] = 0;
            i32x4v gz = {0, 0, 0, 0};
            i32x8v gz8 = {0, 0, 0, 0, 0, 0, 0, 0};
            __builtin_amdgcn_tensor_load_to_lds(g0, g1, gz, gz, gz8, 0);
        }
#else
        for (int idx = tid; idx < 16 * 32; idx += 128) {
            int m = idx >> 5, k = idx & 31;
            sA[idx] = wb[(long)(coutBase + m) * Ktot + kk + k];
        }
#endif
        // --- im2col activation tile 32(k) x 128(n), stored [n][k] ---
        for (int idx = tid; idx < 128 * 32; idx += 128) {
            int n = idx >> 5, k = idx & 31;
            int K = kk + k, v = voxBase + n;
            int ci = K / 27, r = K % 27;
            int kd = r / 9 - 1, kh = (r / 3) % 3 - 1, kw = r % 3 - 1;
            int z = v / (Din * Din), y = (v / Din) % Din, x = v % Din;
            int zz = z + kd, yy = y + kh, xx = x + kw;
            u16 val = 0;
            if ((unsigned)zz < (unsigned)Din && (unsigned)yy < (unsigned)Din &&
                (unsigned)xx < (unsigned)Din) {
                const u16* src = (ci < cin1)
                    ? in1 + ((long)b * cin1 + ci) * numVox
                    : in2 + ((long)b * cin2 + (ci - cin1)) * numVox;
                val = src[(zz * Din + yy) * Din + xx];
            }
            sBt[n * 32 + k] = val;
        }
#if USE_TDM
        __builtin_amdgcn_s_wait_tensorcnt(0);
#endif
        __syncthreads();

        // fragments: two 16B LDS loads each (contiguous K runs per ISA 7.12.2)
        union { v16bf v; u32x4v q[2]; } af, b0, b1;
        af.q[0] = *(const u32x4v*)&sA[l15 * 32 + half];
        af.q[1] = *(const u32x4v*)&sA[l15 * 32 + half + 16];
        int n0 = wv * 32 + l15;
        b0.q[0] = *(const u32x4v*)&sBt[n0 * 32 + half];
        b0.q[1] = *(const u32x4v*)&sBt[n0 * 32 + half + 16];
        int n1 = n0 + 16;
        b1.q[0] = *(const u32x4v*)&sBt[n1 * 32 + half];
        b1.q[1] = *(const u32x4v*)&sBt[n1 * 32 + half + 16];

        acc0 = __builtin_amdgcn_wmma_f32_16x16x32_bf16(
            false, af.v, false, b0.v, (short)0, acc0, false, false);
        acc1 = __builtin_amdgcn_wmma_f32_16x16x32_bf16(
            false, af.v, false, b1.v, (short)0, acc1, false, false);
        __syncthreads();
    }

    // writeback: VGPR i -> M = i + (lane>=16 ? 8 : 0); N = col within wave tile
    int nc = voxBase + wv * 32 + l15;
#pragma unroll
    for (int i = 0; i < 8; ++i) {
        int co = coutBase + i + ((lane >= 16) ? 8 : 0);
        float bs = bias[co];
        long o = ((long)b * Cout + co) * numVox + nc;
        out[o] = acc0[i] + bs;
        out[o + 16] = acc1[i] + bs;
    }
}

// ---------------- GroupNorm stats ----------------
__global__ void k_gnstats(const float* __restrict__ buf, float* __restrict__ stats,
                          int C, int V) {
    int bg = blockIdx.x;
    int cpg = C / 8;
    long base = (long)bg * cpg * V;
    long n = (long)cpg * V;
    float s = 0.f, s2 = 0.f;
    for (long i = threadIdx.x; i < n; i += blockDim.x) {
        float v = buf[base + i];
        s += v; s2 += v * v;
    }
    __shared__ float sh[256], sh2[256];
    sh[threadIdx.x] = s; sh2[threadIdx.x] = s2;
    __syncthreads();
    for (int off = 128; off > 0; off >>= 1) {
        if (threadIdx.x < off) { sh[threadIdx.x] += sh[threadIdx.x + off];
                                 sh2[threadIdx.x] += sh2[threadIdx.x + off]; }
        __syncthreads();
    }
    if (threadIdx.x == 0) {
        float mean = sh[0] / (float)n;
        float var = sh2[0] / (float)n - mean * mean;
        stats[bg * 2] = mean;
        stats[bg * 2 + 1] = rsqrtf(var + 1e-5f);
    }
}

// ---------------- GN apply + exact GELU -> bf16 ----------------
__global__ void k_gngelu(const float* __restrict__ in, u16* __restrict__ out,
                         const float* __restrict__ gam, const float* __restrict__ bet,
                         const float* __restrict__ stats, int C, int V) {
    long total = (long)NB * C * V;
    long i = (long)blockIdx.x * blockDim.x + threadIdx.x;
    if (i >= total) return;
    int cpg = C / 8;
    int c = (int)((i / V) % C);
    long bg = i / ((long)cpg * V);
    float x = (in[i] - stats[bg * 2]) * stats[bg * 2 + 1] * gam[c] + bet[c];
    out[i] = f2bf(0.5f * x * (1.f + erff(x * 0.70710678118654752f)));
}

// ---------------- 2x avgpool (bf16) ----------------
__global__ void k_pool(const u16* __restrict__ in, u16* __restrict__ out,
                       int C, int Din) {
    int Do = Din / 2;
    long total = (long)NB * C * Do * Do * Do;
    long i = (long)blockIdx.x * blockDim.x + threadIdx.x;
    if (i >= total) return;
    int x = (int)(i % Do), y = (int)((i / Do) % Do), z = (int)((i / ((long)Do * Do)) % Do);
    long bc = i / ((long)Do * Do * Do);
    const u16* p = in + bc * Din * Din * Din;
    float s = 0.f;
    for (int dz = 0; dz < 2; ++dz)
        for (int dy = 0; dy < 2; ++dy)
            for (int dx = 0; dx < 2; ++dx)
                s += bf2f(p[((2 * z + dz) * Din + 2 * y + dy) * Din + 2 * x + dx]);
    out[i] = f2bf(s * 0.125f);
}

// ---------------- 2x trilinear upsample (bf16) ----------------
__global__ void k_up(const u16* __restrict__ in, u16* __restrict__ out,
                     int C, int Din) {
    int Do = Din * 2;
    long total = (long)NB * C * Do * Do * Do;
    long i = (long)blockIdx.x * blockDim.x + threadIdx.x;
    if (i >= total) return;
    int x = (int)(i % Do), y = (int)((i / Do) % Do), z = (int)((i / ((long)Do * Do)) % Do);
    long bc = i / ((long)Do * Do * Do);
    const u16* p = in + bc * Din * Din * Din;
    int i0[3], i1[3]; float w1[3];
    int crd[3] = {z, y, x};
    for (int a = 0; a < 3; ++a) {
        float s = 0.5f * (float)crd[a] - 0.25f;
        float f = floorf(s);
        w1[a] = s - f;
        int ii = (int)f;
        i0[a] = min(max(ii, 0), Din - 1);
        i1[a] = min(max(ii + 1, 0), Din - 1);
    }
    float acc = 0.f;
    for (int dz = 0; dz < 2; ++dz)
        for (int dy = 0; dy < 2; ++dy)
            for (int dx = 0; dx < 2; ++dx) {
                float wt = (dz ? w1[0] : 1.f - w1[0]) * (dy ? w1[1] : 1.f - w1[1]) *
                           (dx ? w1[2] : 1.f - w1[2]);
                acc += wt * bf2f(p[((dz ? i1[0] : i0[0]) * Din + (dy ? i1[1] : i0[1])) * Din +
                                   (dx ? i1[2] : i0[2])]);
            }
    out[i] = f2bf(acc);
}

// ---------------- 1x1 out conv (32 -> 64), f32 out ----------------
__global__ void k_out1x1(const u16* __restrict__ in, const float* __restrict__ w,
                         const float* __restrict__ b, float* __restrict__ out) {
    long i = (long)blockIdx.x * blockDim.x + threadIdx.x;
    if (i >= (long)NB * 64 * V32) return;
    int v = (int)(i % V32), o = (int)((i / V32) % 64), bb = (int)(i / (64L * V32));
    float s = b[o];
    const u16* p = in + (long)bb * 32 * V32 + v;
    const float* ww = w + o * 32;
    for (int c = 0; c < 32; ++c) s += ww[c] * bf2f(p[(long)c * V32]);
    out[i] = s;
}

// ---------------- trilinear grid sample (border clamp) ----------------
__global__ void k_sample(const float* __restrict__ vox, const float* __restrict__ pos,
                         float* __restrict__ out) {
    long i = (long)blockIdx.x * blockDim.x + threadIdx.x;
    if (i >= (long)NB * NPTS) return;
    int b = (int)(i / NPTS), n = (int)(i % NPTS);
    const float* pp = pos + ((long)b * NPTS + n) * 3;
    int i0[3], i1[3]; float w1[3];
    for (int a = 0; a < 3; ++a) {
        float c = pp[a] * RR - 0.5f;
        float f = floorf(c);
        w1[a] = c - f;
        int ii = (int)f;
        i0[a] = min(max(ii, 0), RR - 1);
        i1[a] = min(max(ii + 1, 0), RR - 1);
    }
    int off[8]; float wt[8];
    int q = 0;
    for (int dz = 0; dz < 2; ++dz)
        for (int dy = 0; dy < 2; ++dy)
            for (int dx = 0; dx < 2; ++dx, ++q) {
                off[q] = ((dz ? i1[0] : i0[0]) * RR + (dy ? i1[1] : i0[1])) * RR +
                         (dx ? i1[2] : i0[2]);
                wt[q] = (dz ? w1[0] : 1.f - w1[0]) * (dy ? w1[1] : 1.f - w1[1]) *
                        (dx ? w1[2] : 1.f - w1[2]);
            }
    const float* vb = vox + (long)b * 64 * V32;
    float* ob = out + ((long)b * NPTS + n) * 64;
    for (int c = 0; c < 64; ++c) {
        const float* vc = vb + (long)c * V32;
        float s = 0.f;
        for (int j = 0; j < 8; ++j) s += wt[j] * vc[off[j]];
        ob[c] = s;
    }
}

// ======================= host launch =======================
struct Blk { const float *w1,*b1,*g1s,*g1b,*w2,*b2,*g2s,*g2b; };

static void gn(const float* in, u16* outb, const float* gam, const float* bet,
               int C, long V, float* stats, hipStream_t s) {
    k_gnstats<<<NB * 8, 256, 0, s>>>(in, stats, C, (int)V);
    long tot = (long)NB * C * V;
    k_gngelu<<<(unsigned)((tot + 255) / 256), 256, 0, s>>>(in, outb, gam, bet, stats, C, (int)V);
}

extern "C" void kernel_launch(void* const* d_in, const int* in_sizes, int n_in,
                              void* d_out, int out_size, void* d_ws, size_t ws_size,
                              hipStream_t stream) {
    (void)in_sizes; (void)n_in; (void)out_size; (void)ws_size;
    int pi = 0;
    auto F = [&]() { return (const float*)d_in[pi++]; };
    const float* feats = F();
    const float* pos = F();
    const float* widths = F();
    const float* fuse_w = F();
    const float* fuse_b = F();
    auto getblk = [&]() {
        Blk b; b.w1=F(); b.b1=F(); b.g1s=F(); b.g1b=F(); b.w2=F(); b.b2=F(); b.g2s=F(); b.g2b=F();
        return b;
    };
    Blk enc1 = getblk(), enc2 = getblk(), enc3 = getblk(), dec2 = getblk(), dec1 = getblk();
    const float* out_w = F();
    const float* out_b = F();

    // ---- workspace layout: f32 regions first, then bf16 (u16) regions ----
    float* pf = (float*)d_ws;
    float* voxsum = pf; pf += (long)NB * 64 * V32;
    float* cnt    = pf; pf += (long)NB * V32;
    float* ctx    = pf; pf += (long)NB * 7 * V32;
    float* t32f   = pf; pf += (long)NB * 32 * V32;   // pre-GN f32 temp @32^3
    float* t16f   = pf; pf += (long)NB * 64 * V16;   // @16^3
    float* t8f    = pf; pf += (long)NB * 128 * V8;   // @8^3
    float* vout   = pf; pf += (long)NB * 64 * V32;
    float* stats  = pf; pf += 64;
    u16* pu = (u16*)pf;
    u16* x0   = pu; pu += (long)NB * 64 * V32;
    u16* a32  = pu; pu += (long)NB * 32 * V32;       // act temp @32^3
    u16* e1   = pu; pu += (long)NB * 32 * V32;
    u16* p1   = pu; pu += (long)NB * 32 * V16;
    u16* a16  = pu; pu += (long)NB * 64 * V16;
    u16* e2   = pu; pu += (long)NB * 64 * V16;
    u16* p2   = pu; pu += (long)NB * 64 * V8;
    u16* a8   = pu; pu += (long)NB * 128 * V8;
    u16* e3   = pu; pu += (long)NB * 128 * V8;
    u16* u3   = pu; pu += (long)NB * 128 * V16;
    u16* d2   = pu; pu += (long)NB * 64 * V16;
    u16* u2   = pu; pu += (long)NB * 64 * V32;
    u16* d1   = pu; pu += (long)NB * 32 * V32;
    // bf16 weight matrices [Cout][Cin*27]
    const long wn[10] = {55296, 27648, 55296, 110592, 221184, 442368, 331776, 110592, 82944, 27648};
    const float* wsrc[10] = {enc1.w1, enc1.w2, enc2.w1, enc2.w2, enc3.w1,
                             enc3.w2, dec2.w1, dec2.w2, dec1.w1, dec1.w2};
    u16* wdst[10];
    for (int j = 0; j < 10; ++j) { wdst[j] = pu; pu += wn[j]; }

    // 0) convert conv weights to bf16 GEMM layout
    for (int j = 0; j < 10; ++j)
        k_cvt<<<(unsigned)((wn[j] + 255) / 256), 256, 0, stream>>>(wsrc[j], wdst[j], wn[j]);

    // 1) scatter-mean accumulators
    long nz = (long)NB * 65 * V32;   // voxsum + cnt contiguous
    k_zero<<<(unsigned)((nz + 255) / 256), 256, 0, stream>>>(voxsum, nz);
    k_scatter<<<(NB * NPTS + 255) / 256, 256, 0, stream>>>(feats, pos, voxsum, cnt);

    // 2) context + fuse -> bf16 x0
    k_ctx<<<NB, 256, 0, stream>>>(widths, ctx);
    long nf = (long)NB * 64 * V32;
    k_fuse<<<(unsigned)((nf + 255) / 256), 256, 0, stream>>>(voxsum, cnt, ctx, fuse_w, fuse_b, x0);

    // 3) enc1 @32^3 (64->32, 32->32)
    k_conv3<<<dim3(V32 / 128, 2, NB), 128, 0, stream>>>(x0, nullptr, 64, 0, wdst[0], enc1.b1, t32f, 32, 32);
    gn(t32f, a32, enc1.g1s, enc1.g1b, 32, V32, stats, stream);
    k_conv3<<<dim3(V32 / 128, 2, NB), 128, 0, stream>>>(a32, nullptr, 32, 0, wdst[1], enc1.b2, t32f, 32, 32);
    gn(t32f, e1, enc1.g2s, enc1.g2b, 32, V32, stats, stream);

    // 4) enc2 @16^3 (32->64, 64->64)
    long np1 = (long)NB * 32 * V16;
    k_pool<<<(unsigned)((np1 + 255) / 256), 256, 0, stream>>>(e1, p1, 32, 32);
    k_conv3<<<dim3(V16 / 128, 4, NB), 128, 0, stream>>>(p1, nullptr, 32, 0, wdst[2], enc2.b1, t16f, 16, 64);
    gn(t16f, a16, enc2.g1s, enc2.g1b, 64, V16, stats, stream);
    k_conv3<<<dim3(V16 / 128, 4, NB), 128, 0, stream>>>(a16, nullptr, 64, 0, wdst[3], enc2.b2, t16f, 16, 64);
    gn(t16f, e2, enc2.g2s, enc2.g2b, 64, V16, stats, stream);

    // 5) enc3 @8^3 (64->128, 128->128)
    long np2 = (long)NB * 64 * V8;
    k_pool<<<(unsigned)((np2 + 255) / 256), 256, 0, stream>>>(e2, p2, 64, 16);
    k_conv3<<<dim3(V8 / 128, 8, NB), 128, 0, stream>>>(p2, nullptr, 64, 0, wdst[4], enc3.b1, t8f, 8, 128);
    gn(t8f, a8, enc3.g1s, enc3.g1b, 128, V8, stats, stream);
    k_conv3<<<dim3(V8 / 128, 8, NB), 128, 0, stream>>>(a8, nullptr, 128, 0, wdst[5], enc3.b2, t8f, 8, 128);
    gn(t8f, e3, enc3.g2s, enc3.g2b, 128, V8, stats, stream);

    // 6) dec2 @16^3: concat(up2(e3)[128], e2[64]) -> 64, then 64->64
    long nu3 = (long)NB * 128 * V16;
    k_up<<<(unsigned)((nu3 + 255) / 256), 256, 0, stream>>>(e3, u3, 128, 8);
    k_conv3<<<dim3(V16 / 128, 4, NB), 128, 0, stream>>>(u3, e2, 128, 64, wdst[6], dec2.b1, t16f, 16, 64);
    gn(t16f, a16, dec2.g1s, dec2.g1b, 64, V16, stats, stream);
    k_conv3<<<dim3(V16 / 128, 4, NB), 128, 0, stream>>>(a16, nullptr, 64, 0, wdst[7], dec2.b2, t16f, 16, 64);
    gn(t16f, d2, dec2.g2s, dec2.g2b, 64, V16, stats, stream);

    // 7) dec1 @32^3: concat(up2(d2)[64], e1[32]) -> 32, then 32->32
    long nu2 = (long)NB * 64 * V32;
    k_up<<<(unsigned)((nu2 + 255) / 256), 256, 0, stream>>>(d2, u2, 64, 16);
    k_conv3<<<dim3(V32 / 128, 2, NB), 128, 0, stream>>>(u2, e1, 64, 32, wdst[8], dec1.b1, t32f, 32, 32);
    gn(t32f, a32, dec1.g1s, dec1.g1b, 32, V32, stats, stream);
    k_conv3<<<dim3(V32 / 128, 2, NB), 128, 0, stream>>>(a32, nullptr, 32, 0, wdst[9], dec1.b2, t32f, 32, 32);
    gn(t32f, d1, dec1.g2s, dec1.g2b, 32, V32, stats, stream);

    // 8) 1x1 out conv (32 -> 64)
    long no = (long)NB * 64 * V32;
    k_out1x1<<<(unsigned)((no + 255) / 256), 256, 0, stream>>>(d1, out_w, out_b, vout);

    // 9) trilinear sample back to points -> [B,N,64]
    k_sample<<<(NB * NPTS + 255) / 256, 256, 0, stream>>>(vout, pos, (float*)d_out);
}
